// Attention_14181982011875
// MI455X (gfx1250) — matrix operand order repaired and checked by reading, not compiled, not run on hardware
//
#include <hip/hip_runtime.h>

// ---------------------------------------------------------------------------
// CDNA5 (gfx1250, wave32) attention layer built on v_wmma_f32_16x16x32_bf16
// with GLOBAL_LOAD_ASYNC_TO_LDS_B128 double-buffered staging (ASYNCcnt).
//   HIDDEN=2048, N_HEADS=16, N_KV_HEADS=4, HEAD_DIM=128, BATCH=2, SEQ=2048
// ---------------------------------------------------------------------------

typedef __bf16 bf16x16 __attribute__((ext_vector_type(16)));
typedef float  f32x8   __attribute__((ext_vector_type(8)));

union Frag16 {            // A/B operand fragment: 16 bf16 per lane (8 VGPRs)
    bf16x16      v;
    unsigned int u[8];
};
union Frag32 {            // C/D accumulator fragment: 8 f32 per lane
    f32x8 v;
    float f[8];
};

__device__ __forceinline__ unsigned short f2bf(float f) {
    unsigned int u = __float_as_uint(f);
    u += 0x7FFFu + ((u >> 16) & 1u);            // round-to-nearest-even
    return (unsigned short)(u >> 16);
}

// ---- CDNA5 async global->LDS DMA (ISA 15.18.3 op 98; ASYNCcnt) -------------
// lds = wave-relative LDS byte address (low 32 bits of generic pointer to
// __shared__, per ISA 10.2 aperture rules).  IOFFSET applies to BOTH the LDS
// and the global address, so one base pair serves a 64B strip via offsets.
#define ASYNC_G2L_B128(lds, ga, off_)                                          \
    asm volatile("global_load_async_to_lds_b128 %0, %1, off offset:" off_      \
                 :: "v"(lds), "v"(ga) : "memory")

#define S_WAIT_ASYNCCNT(n_)                                                    \
    asm volatile("s_wait_asynccnt " n_ ::: "memory")

__device__ __forceinline__ unsigned lds_addr(const void* p) {
    return (unsigned)(size_t)p;
}

// ---------------------------------------------------------------------------
// elementwise f32 -> bf16 (for x, which stays row-major [M][K])
// ---------------------------------------------------------------------------
__global__ void cvt_f32_bf16(const float* __restrict__ src,
                             unsigned short* __restrict__ dst, int n) {
    int i = blockIdx.x * blockDim.x + threadIdx.x;
    if (i < n) dst[i] = f2bf(src[i]);
}

// ---------------------------------------------------------------------------
// transpose + convert: w[K][N] f32  ->  wT[N][K] bf16  (done once; weights are
// reused by 32 row-blocks, so GEMM staging becomes contiguous b128 DMA)
// ---------------------------------------------------------------------------
__global__ __launch_bounds__(256) void cvt_transpose_f32_bf16(
    const float* __restrict__ src, unsigned short* __restrict__ dst,
    int K, int N) {
    __shared__ float tile[32][33];
    const int kx = blockIdx.x * 32;
    const int nx = blockIdx.y * 32;
    const int tx = threadIdx.x;           // 0..31
#pragma unroll
    for (int i = threadIdx.y; i < 32; i += 8)
        tile[i][tx] = src[(size_t)(kx + i) * N + nx + tx];
    __syncthreads();
#pragma unroll
    for (int i = threadIdx.y; i < 32; i += 8)
        dst[(size_t)(nx + i) * K + kx + tx] = f2bf(tile[tx][i]);
}

// ---------------------------------------------------------------------------
// Async double-buffered bf16 GEMM: C[M,N] = A[M,K] * BT[N,K]^T (+bias).
// 128x64 tile per block of 4 wave32 waves; K stepped by 32.
// Tile i+1/i+2 stream into LDS via async DMA while tile i computes.
// OUT_MODE: 0 = bf16 row-major, 1 = f32 row-major,
//           2 = bf16 transposed for V: [(row>>11)*512 + col][2048] at (row&2047)
// ---------------------------------------------------------------------------
template <int OUT_MODE>
__global__ __launch_bounds__(128) void gemm_bf16(
    const unsigned short* __restrict__ A,   // [M,K] bf16 bits
    const unsigned short* __restrict__ BT,  // [N,K] bf16 bits (pre-transposed)
    const float* __restrict__ bias,         // [N] or nullptr
    void* __restrict__ Cout,
    int M, int N, int K) {
    __shared__ unsigned short At[2][128 * 32]; // [m][k] per buffer
    __shared__ unsigned short Bt[2][64 * 32];  // [n][k] per buffer

    const int tid  = threadIdx.x;
    const int lane = tid & 31;
    const int wave = tid >> 5;
    const int half = (lane >> 4) & 1;
    const int r    = lane & 15;
    const int m0   = blockIdx.y * 128;
    const int n0   = blockIdx.x * 64;
    const int mw   = wave * 32;             // this wave's 32-row strip

    Frag32 acc[2][4];
#pragma unroll
    for (int s = 0; s < 2; s++)
#pragma unroll
        for (int t = 0; t < 4; t++)
#pragma unroll
            for (int j = 0; j < 8; j++) acc[s][t].f[j] = 0.0f;

    // per-thread DMA strips: A row (m0+tid) 64B, B row (n0+tid/2) 32B
    const unsigned short* aRow = A + (size_t)(m0 + tid) * K;
    const unsigned short* bRow = BT + (size_t)(n0 + (tid >> 1)) * K + ((tid & 1) << 4);
    const unsigned ldsA0 = lds_addr(&At[0][tid * 32]);
    const unsigned ldsA1 = lds_addr(&At[1][tid * 32]);
    const unsigned ldsB0 = lds_addr(&Bt[0][(tid >> 1) * 32 + ((tid & 1) << 4)]);
    const unsigned ldsB1 = lds_addr(&Bt[1][(tid >> 1) * 32 + ((tid & 1) << 4)]);

    auto issue = [&](int it, int buf) {     // 6 async ops per tile per thread
        unsigned long long gaA = (unsigned long long)(size_t)(aRow + it * 32);
        unsigned long long gaB = (unsigned long long)(size_t)(bRow + it * 32);
        unsigned la = buf ? ldsA1 : ldsA0;
        unsigned lb = buf ? ldsB1 : ldsB0;
        ASYNC_G2L_B128(la, gaA, "0");
        ASYNC_G2L_B128(la, gaA, "16");
        ASYNC_G2L_B128(la, gaA, "32");
        ASYNC_G2L_B128(la, gaA, "48");
        ASYNC_G2L_B128(lb, gaB, "0");
        ASYNC_G2L_B128(lb, gaB, "16");
    };

    const int NIT = K >> 5;                 // k-steps of 32
    issue(0, 0);
    if (NIT > 1) issue(1, 1);

    for (int it = 0; it < NIT; ++it) {
        // tile 'it' complete; allow tile it+1's 6 ops to stay in flight
        if (it + 1 < NIT) { S_WAIT_ASYNCCNT("0x6"); }
        else              { S_WAIT_ASYNCCNT("0x0"); }
        __syncthreads();

        const unsigned short* Ac = At[it & 1];
        const unsigned short* Bc = Bt[it & 1];
        Frag16 a[2];
#pragma unroll
        for (int s = 0; s < 2; s++)
#pragma unroll
            for (int p = 0; p < 8; p++) {
                int kk = 2 * p + (half ? 8 : 0) + (p >= 4 ? 8 : 0);
                a[s].u[p] = *(const unsigned int*)&Ac[(mw + s * 16 + r) * 32 + kk];
            }
#pragma unroll
        for (int t = 0; t < 4; t++) {
            Frag16 b;
#pragma unroll
            for (int p = 0; p < 8; p++) {
                int kk = 2 * p + (half ? 8 : 0) + (p >= 4 ? 8 : 0);
                b.u[p] = *(const unsigned int*)&Bc[(t * 16 + r) * 32 + kk];
            }
            acc[0][t].v = __builtin_amdgcn_wmma_f32_16x16x32_bf16(
                false, a[0].v, false, b.v, (short)0, acc[0][t].v, false, false);
            acc[1][t].v = __builtin_amdgcn_wmma_f32_16x16x32_bf16(
                false, a[1].v, false, b.v, (short)0, acc[1][t].v, false, false);
        }
        __syncthreads();                    // all reads of this buffer done
        if (it + 2 < NIT) issue(it + 2, it & 1);
    }

#pragma unroll
    for (int s = 0; s < 2; s++)
#pragma unroll
        for (int t = 0; t < 4; t++) {
            int col  = n0 + t * 16 + r;
            float bs = bias ? bias[col] : 0.0f;
#pragma unroll
            for (int j = 0; j < 8; j++) {
                int row   = m0 + mw + s * 16 + j + half * 8;
                float val = acc[s][t].f[j] + bs;
                if (OUT_MODE == 1) {
                    ((float*)Cout)[(size_t)row * N + col] = val;
                } else if (OUT_MODE == 0) {
                    ((unsigned short*)Cout)[(size_t)row * N + col] = f2bf(val);
                } else {  // V transposed: [b][col][s], b = row>>11, s = row&2047
                    ((unsigned short*)Cout)[((size_t)(row >> 11) * 512 + col) * 2048 +
                                            (row & 2047)] = f2bf(val);
                }
            }
        }
}

// ---------------------------------------------------------------------------
// Flash attention (causal, GQA 4:1).  Block = 4 waves, one (b, h, 64-q-rows)
// tile; each wave owns a 16-row Q strip.  32-key steps, K/V^T tiles stream
// into double-buffered LDS via async DMA one step ahead of compute.
//   S(16x32)  = Q(16x128) . K^T        -> 8 WMMA
//   online softmax (shfl_xor row reductions in C-fragment layout)
//   O(16x128) += P(16x32) . V(32x128)  -> 8 WMMA
// ---------------------------------------------------------------------------
__global__ __launch_bounds__(128) void attn_flash(
    const unsigned short* __restrict__ Qb,  // [B,S,16*128]
    const unsigned short* __restrict__ Kb,  // [B,S,4*128]
    const unsigned short* __restrict__ VT,  // [B, 4*128, S]  (pre-transposed)
    unsigned short* __restrict__ Ob) {      // [B,S,16*128]
    __shared__ unsigned short Kt[2][32 * 128];   // [key][d]  (B-frag for QK^T)
    __shared__ unsigned short Vt[2][128 * 32];   // [d][key]  (B-frag for PV)
    __shared__ unsigned short Pb[4][16 * 32];    // per-wave P scratch [q][key]

    const int tid  = threadIdx.x;
    const int lane = tid & 31;
    const int wave = tid >> 5;
    const int half = (lane >> 4) & 1;
    const int r    = lane & 15;

    const int qblock = blockIdx.x;  // 0..31
    const int h      = blockIdx.y;  // 0..15
    const int b      = blockIdx.z;  // 0..1
    const int kvh    = h >> 2;      // GQA: kv head = h / 4
    const int q0     = qblock * 64 + wave * 16;
    const float scale = 0.08838834764831845f;  // 1/sqrt(128)

    // per-thread DMA strips
    const int krow  = tid >> 2;        // 0..31
    const int dpart = (tid & 3) * 32;  // 0,32,64,96
    const unsigned short* kSrc =
        Kb + ((size_t)(b * 2048) + krow) * 512 + kvh * 128 + dpart;
    const unsigned short* vSrc =
        VT + ((size_t)(b * 512) + kvh * 128 + tid) * 2048;   // d = tid
    const unsigned ldsK0 = lds_addr(&Kt[0][krow * 128 + dpart]);
    const unsigned ldsK1 = lds_addr(&Kt[1][krow * 128 + dpart]);
    const unsigned ldsV0 = lds_addr(&Vt[0][tid * 32]);
    const unsigned ldsV1 = lds_addr(&Vt[1][tid * 32]);

    auto issue = [&](int kb, int buf) {    // 8 async ops per tile per thread
        unsigned long long gaK =
            (unsigned long long)(size_t)(kSrc + (size_t)(kb * 32) * 512);
        unsigned long long gaV =
            (unsigned long long)(size_t)(vSrc + kb * 32);
        unsigned lk = buf ? ldsK1 : ldsK0;
        unsigned lv = buf ? ldsV1 : ldsV0;
        ASYNC_G2L_B128(lk, gaK, "0");
        ASYNC_G2L_B128(lk, gaK, "16");
        ASYNC_G2L_B128(lk, gaK, "32");
        ASYNC_G2L_B128(lk, gaK, "48");
        ASYNC_G2L_B128(lv, gaV, "0");
        ASYNC_G2L_B128(lv, gaV, "16");
        ASYNC_G2L_B128(lv, gaV, "32");
        ASYNC_G2L_B128(lv, gaV, "48");
    };

    // Resident Q fragments (4 x 32-wide K chunks over HEAD_DIM=128)
    Frag16 qf[4];
    const unsigned short* qrow =
        Qb + ((size_t)(b * 2048) + q0 + r) * 2048 + h * 128;
#pragma unroll
    for (int c = 0; c < 4; c++)
#pragma unroll
        for (int p = 0; p < 8; p++) {
            int kk = c * 32 + 2 * p + (half ? 8 : 0) + (p >= 4 ? 8 : 0);
            qf[c].u[p] = *(const unsigned int*)(qrow + kk);
        }

    Frag32 o[8];
#pragma unroll
    for (int t = 0; t < 8; t++)
#pragma unroll
        for (int j = 0; j < 8; j++) o[t].f[j] = 0.0f;

    float mrow[8], lrow[8];
#pragma unroll
    for (int j = 0; j < 8; j++) { mrow[j] = -3.0e38f; lrow[j] = 0.0f; }

    const int nkb = qblock * 2 + 2;  // causal: keys up to q-block end
    issue(0, 0);
    if (nkb > 1) issue(1, 1);

    for (int kb = 0; kb < nkb; kb++) {
        const int ks0 = kb * 32;
        if (kb + 1 < nkb) { S_WAIT_ASYNCCNT("0x8"); }
        else              { S_WAIT_ASYNCCNT("0x0"); }
        __syncthreads();
        const unsigned short* Kc = Kt[kb & 1];
        const unsigned short* Vc = Vt[kb & 1];

        // ---- S = Q . K^T  (two 16-key column tiles) ----
        Frag32 s[2];
#pragma unroll
        for (int t = 0; t < 2; t++) {
#pragma unroll
            for (int j = 0; j < 8; j++) s[t].f[j] = 0.0f;
#pragma unroll
            for (int c = 0; c < 4; c++) {
                Frag16 kf;
#pragma unroll
                for (int p = 0; p < 8; p++) {
                    int dd = c * 32 + 2 * p + (half ? 8 : 0) + (p >= 4 ? 8 : 0);
                    kf.u[p] = *(const unsigned int*)&Kc[(t * 16 + r) * 128 + dd];
                }
                s[t].v = __builtin_amdgcn_wmma_f32_16x16x32_bf16(
                    false, qf[c].v, false, kf.v, (short)0, s[t].v, false, false);
            }
        }

        // ---- online softmax (rows = j + half*8 of this wave's q strip) ----
        float p0[8], p1[8], alpha[8];
#pragma unroll
        for (int j = 0; j < 8; j++) {
            int q    = q0 + j + half * 8;
            int key0 = ks0 + r;
            int key1 = ks0 + 16 + r;
            float v0 = (key0 <= q) ? s[0].f[j] * scale : -1.0e30f;
            float v1 = (key1 <= q) ? s[1].f[j] * scale : -1.0e30f;
            float vm = fmaxf(v0, v1);
            vm = fmaxf(vm, __shfl_xor(vm, 1, 32));
            vm = fmaxf(vm, __shfl_xor(vm, 2, 32));
            vm = fmaxf(vm, __shfl_xor(vm, 4, 32));
            vm = fmaxf(vm, __shfl_xor(vm, 8, 32));
            float mnew = fmaxf(mrow[j], vm);
            alpha[j] = __expf(mrow[j] - mnew);
            mrow[j]  = mnew;
            p0[j] = __expf(v0 - mnew);
            p1[j] = __expf(v1 - mnew);
            float ps = p0[j] + p1[j];
            ps += __shfl_xor(ps, 1, 32);
            ps += __shfl_xor(ps, 2, 32);
            ps += __shfl_xor(ps, 4, 32);
            ps += __shfl_xor(ps, 8, 32);
            lrow[j] = lrow[j] * alpha[j] + ps;
        }

        // rescale O, park P in per-wave LDS to re-fragment as an A matrix
#pragma unroll
        for (int j = 0; j < 8; j++) {
            Pb[wave][(j + half * 8) * 32 + r]      = f2bf(p0[j]);
            Pb[wave][(j + half * 8) * 32 + 16 + r] = f2bf(p1[j]);
#pragma unroll
            for (int t2 = 0; t2 < 8; t2++) o[t2].f[j] *= alpha[j];
        }

        Frag16 pf;
#pragma unroll
        for (int p = 0; p < 8; p++) {
            int kk = 2 * p + (half ? 8 : 0) + (p >= 4 ? 8 : 0);
            pf.u[p] = *(const unsigned int*)&Pb[wave][r * 32 + kk];
        }

        // ---- O += P . V  (8 d-column tiles) ----
#pragma unroll
        for (int t2 = 0; t2 < 8; t2++) {
            Frag16 vf;
#pragma unroll
            for (int p = 0; p < 8; p++) {
                int kk = 2 * p + (half ? 8 : 0) + (p >= 4 ? 8 : 0);
                vf.u[p] = *(const unsigned int*)&Vc[(t2 * 16 + r) * 32 + kk];
            }
            o[t2].v = __builtin_amdgcn_wmma_f32_16x16x32_bf16(
                false, pf.v, false, vf.v, (short)0, o[t2].v, false, false);
        }

        __syncthreads();                    // done reading buffer kb&1
        if (kb + 2 < nkb) issue(kb + 2, kb & 1);
    }

    // ---- normalize and store O as bf16 [b, q, h*128 + d] ----
#pragma unroll
    for (int j = 0; j < 8; j++) {
        float inv  = 1.0f / lrow[j];
        size_t row = (size_t)(b * 2048) + q0 + j + half * 8;
#pragma unroll
        for (int t2 = 0; t2 < 8; t2++)
            Ob[row * 2048 + h * 128 + t2 * 16 + r] = f2bf(o[t2].f[j] * inv);
    }
}

// ---------------------------------------------------------------------------
// Host-side orchestration
// ---------------------------------------------------------------------------
extern "C" void kernel_launch(void* const* d_in, const int* in_sizes, int n_in,
                              void* d_out, int out_size, void* d_ws,
                              size_t ws_size, hipStream_t stream) {
    (void)in_sizes; (void)n_in; (void)out_size; (void)ws_size;

    const int HIDDEN = 2048, BATCH = 2, SEQ = 2048;
    const int M = BATCH * SEQ;          // 4096
    const int NKV = 512;                // 4 * 128

    const float* x  = (const float*)d_in[0];
    // d_in[1] = attention_mask (causal handled analytically), d_in[2] = position_ids (unused)
    const float* wq = (const float*)d_in[3];
    const float* bq = (const float*)d_in[4];
    const float* wk = (const float*)d_in[5];
    const float* bk = (const float*)d_in[6];
    const float* wv = (const float*)d_in[7];
    const float* bv = (const float*)d_in[8];
    const float* wo = (const float*)d_in[9];

    // workspace layout (bf16 bit buffers)
    char* ws = (char*)d_ws;
    size_t off = 0;
    unsigned short* xb  = (unsigned short*)(ws + off); off += (size_t)M * HIDDEN * 2;
    unsigned short* wqT = (unsigned short*)(ws + off); off += (size_t)HIDDEN * HIDDEN * 2;
    unsigned short* wkT = (unsigned short*)(ws + off); off += (size_t)HIDDEN * NKV * 2;
    unsigned short* wvT = (unsigned short*)(ws + off); off += (size_t)HIDDEN * NKV * 2;
    unsigned short* woT = (unsigned short*)(ws + off); off += (size_t)HIDDEN * HIDDEN * 2;
    unsigned short* Qb  = (unsigned short*)(ws + off); off += (size_t)M * HIDDEN * 2;
    unsigned short* Kb  = (unsigned short*)(ws + off); off += (size_t)M * NKV * 2;
    unsigned short* VT  = (unsigned short*)(ws + off); off += (size_t)M * NKV * 2;
    unsigned short* Ab  = (unsigned short*)(ws + off); off += (size_t)M * HIDDEN * 2;

    // x: elementwise convert (stays row-major)
    cvt_f32_bf16<<<(M * HIDDEN + 255) / 256, 256, 0, stream>>>(x, xb, M * HIDDEN);
    // weights: transpose-convert once -> [N][K]
    cvt_transpose_f32_bf16<<<dim3(HIDDEN / 32, HIDDEN / 32), dim3(32, 8), 0, stream>>>(
        wq, wqT, HIDDEN, HIDDEN);
    cvt_transpose_f32_bf16<<<dim3(HIDDEN / 32, NKV / 32), dim3(32, 8), 0, stream>>>(
        wk, wkT, HIDDEN, NKV);
    cvt_transpose_f32_bf16<<<dim3(HIDDEN / 32, NKV / 32), dim3(32, 8), 0, stream>>>(
        wv, wvT, HIDDEN, NKV);
    cvt_transpose_f32_bf16<<<dim3(HIDDEN / 32, HIDDEN / 32), dim3(32, 8), 0, stream>>>(
        wo, woT, HIDDEN, HIDDEN);

    // Q/K/V projections (bf16 out, fused bias); V written transposed for PV
    gemm_bf16<0><<<dim3(HIDDEN / 64, M / 128), 128, 0, stream>>>(
        xb, wqT, bq, Qb, M, HIDDEN, HIDDEN);
    gemm_bf16<0><<<dim3(NKV / 64, M / 128), 128, 0, stream>>>(
        xb, wkT, bk, Kb, M, NKV, HIDDEN);
    gemm_bf16<2><<<dim3(NKV / 64, M / 128), 128, 0, stream>>>(
        xb, wvT, bv, VT, M, NKV, HIDDEN);

    // flash attention: grid = (q blocks of 64, heads, batch)
    attn_flash<<<dim3(SEQ / 64, 16, BATCH), 128, 0, stream>>>(Qb, Kb, VT, Ab);

    // output projection -> f32 d_out
    gemm_bf16<1><<<dim3(HIDDEN / 64, M / 128), 128, 0, stream>>>(
        Ab, woT, nullptr, d_out, M, HIDDEN, HIDDEN);
}
